// FATM_28870770164494
// MI455X (gfx1250) — matrix operand order repaired and checked by reading, not compiled, not run on hardware
//
#include <hip/hip_runtime.h>
#include <hip/hip_bf16.h>

typedef _Float16 half_t;
typedef __attribute__((ext_vector_type(16))) _Float16 v16h;
typedef __attribute__((ext_vector_type(8)))  _Float16 v8h;
typedef __attribute__((ext_vector_type(8)))  float    v8f;
typedef __attribute__((ext_vector_type(4)))  float    v4f;
typedef __attribute__((ext_vector_type(4)))  int      v4i;
typedef __attribute__((ext_vector_type(8)))  int      v8i;
typedef __attribute__((ext_vector_type(4)))  unsigned int v4u;

#define T_  4
#define B_  8
#define C_  256
#define H_  64
#define W_  64
#define NB_ 16
#define BS_ 16
#define HW_ (H_ * W_)
#define NEL ((size_t)T_ * B_ * C_ * H_ * W_)        // 33,554,432
#define BNCNT 131072.0f                             // T*B*H*W per channel
#define EPS_ 1e-5f

// ---------------- CDNA5 async global->LDS (ASYNCcnt) with safe fallback ------------
// Probe (round 2 stderr): builtin params are int4 pointers with explicit address
// spaces (AS1 prints as '__device__' in HIP); 4-arg form passed the arity check.

#if defined(__has_builtin)
#if __has_builtin(__builtin_amdgcn_global_load_async_to_lds_b128)
#define HAVE_ASYNC_LDS 1
#endif
#if __has_builtin(__builtin_amdgcn_tensor_load_to_lds) && \
    __has_builtin(__builtin_amdgcn_s_wait_tensorcnt)
#define HAVE_TDM 1
#endif
#endif

__device__ inline void async_ld_b128(void* lds, const void* g) {
#ifdef HAVE_ASYNC_LDS
  typedef __attribute__((address_space(1))) v4i g_v4i;
  typedef __attribute__((address_space(3))) v4i l_v4i;
  __builtin_amdgcn_global_load_async_to_lds_b128(
      (g_v4i*)(uintptr_t)g,
      (l_v4i*)(uintptr_t)(uint32_t)(uintptr_t)lds, 0, 0);
#else
  *(v8h*)lds = *(const v8h*)g;
#endif
}

__device__ inline void async_wait() {
#ifdef HAVE_ASYNC_LDS
#if __has_builtin(__builtin_amdgcn_s_wait_asynccnt)
  __builtin_amdgcn_s_wait_asynccnt(0);
#else
  asm volatile("s_wait_asynccnt 0" ::: "memory");
#endif
#endif
}

// ---------------- CDNA5 Tensor Data Mover: contiguous row of 16-bit elements -------
// D# per CDNA5 ISA: group0 = {count=1, lds_addr, global_addr,
// type=2}; group1 = {data_size=1 (2B), tensor_dim0=tile_dim0=n, dim1=1, stride=n}.

#ifdef HAVE_TDM
__device__ inline void tdm_copy_row_to_lds(void* lds, const void* g, int nelem) {
  uint64_t ga = (uint64_t)(uintptr_t)g;
  v4u g0;
  g0[0] = 1u;                                        // count=1, user descriptor
  g0[1] = (uint32_t)(uintptr_t)lds;                  // lds_addr (low 32 bits)
  g0[2] = (uint32_t)ga;                              // global_addr[31:0]
  g0[3] = ((uint32_t)(ga >> 32) & 0x01FFFFFFu) | 0x80000000u; // ga[56:32]|type=2
  v8i g1;
  g1[0] = 0x00010000;                                // data_size=1 (2 bytes)
  g1[1] = (int)((unsigned)(nelem & 0xFFFF) << 16);   // tensor_dim0[15:0]
  g1[2] = (int)(((unsigned)nelem >> 16) | (1u << 16)); // dim0 hi | tensor_dim1=1
  g1[3] = (int)((unsigned)(nelem & 0xFFFF) << 16);   // tile_dim0
  g1[4] = 1;                                         // tile_dim1=1
  g1[5] = nelem;                                     // tensor_dim0_stride lo
  g1[6] = 0;
  g1[7] = 0;
  v4i z4 = {0, 0, 0, 0};
#if __clang_major__ >= 23
  v8i z8 = {0, 0, 0, 0, 0, 0, 0, 0};
  __builtin_amdgcn_tensor_load_to_lds(g0, g1, z4, z4, z8, 0);
#else
  __builtin_amdgcn_tensor_load_to_lds(g0, g1, z4, z4, 0);
#endif
}
#endif

// ---------------- WMMA helpers (CDNA5 16x16x32 f16 layouts, ISA 7.12.2) ------------

__device__ inline v8f wmma_f16(v16h a, v16h b, v8f c) {
  return __builtin_amdgcn_wmma_f32_16x16x32_f16(false, a, false, b, (short)0, c,
                                                false, false);
}

__device__ inline v16h cat8(v8h lo, v8h hi) {
  v16h r;
#pragma unroll
  for (int j = 0; j < 8; ++j) { r[j] = lo[j]; r[j + 8] = hi[j]; }
  return r;
}

__device__ inline v8h zero8() {
  v8h z;
#pragma unroll
  for (int j = 0; j < 8; ++j) z[j] = (half_t)0.0f;
  return z;
}

// A (16x32) from a row-major matrix (leading dim ld halves): 2x b128 per lane.
__device__ inline v16h load_a_rm(const half_t* base, int ld, int lane) {
  int m = lane & 15;
  int kh = (lane >> 4) ? 16 : 0;
  const half_t* p = base + m * ld + kh;
  v8h lo = *(const v8h*)(p);
  v8h hi = *(const v8h*)(p + 8);
  return cat8(lo, hi);
}

// B (32x16) element (k,n) = baseT[n*ld + k]  (K contiguous per N): 2x b128 per lane.
__device__ inline v16h load_b_kcontig(const half_t* baseT, int ld, int lane) {
  int n = lane & 15;
  int kb = (lane >> 4) ? 16 : 0;
  const half_t* p = baseT + n * ld + kb;
  v8h lo = *(const v8h*)(p);
  v8h hi = *(const v8h*)(p + 8);
  return cat8(lo, hi);
}

// Generic gathered B fragment (only used for stage-1 X, inherently strided)
template <typename F>
__device__ inline v16h make_b_frag(int lane, F f) {
  v16h b;
  int n  = lane & 15;
  int kb = (lane >> 4) ? 16 : 0;
#pragma unroll
  for (int j = 0; j < 16; ++j) b[j] = f(kb + j, n);
  return b;
}

// ---------------- K0: build orthonormal Haar matrix (M and M^T) + zero stats -------

__global__ void __launch_bounds__(256)
haar_init_kernel(half_t* __restrict__ Mg, half_t* __restrict__ MgT,
                 float* __restrict__ stats) {
  __shared__ float Ma[64 * 64];
  __shared__ float Mb[64 * 64];
  int tid = threadIdx.x;
  if (tid == 0) {
    Ma[0] = 1.0f;
    for (int n = 2; n <= 64; n <<= 1) {
      int hf = n >> 1;
      float sq = sqrtf((float)hf);
      for (int r = 0; r < hf; ++r)
        for (int j = 0; j < hf; ++j) {
          float tv = Ma[r * 64 + j];
          Mb[r * 64 + 2 * j]            = tv;
          Mb[r * 64 + 2 * j + 1]        = tv;
          Mb[(hf + r) * 64 + 2 * j]     = (r == j) ? sq : 0.0f;
          Mb[(hf + r) * 64 + 2 * j + 1] = (r == j) ? -sq : 0.0f;
        }
      for (int r = 0; r < n; ++r)
        for (int j = 0; j < n; ++j) Ma[r * 64 + j] = Mb[r * 64 + j];
    }
  }
  __syncthreads();
  for (int i = tid; i < 64 * 64; i += 256) {
    int r = i >> 6, cc = i & 63;
    half_t v = (half_t)(Ma[i] * 0.125f);          // /sqrt(64)
    Mg[i] = v;
    MgT[cc * 64 + r] = v;
  }
  for (int i = tid; i < 5 * 512; i += 256) stats[i] = 0.0f;
}

// ---------------- K1: LIF over T, f32 -> f16 spikes (8 elems/thread, b128 I/O) -----

__global__ void __launch_bounds__(256)
lif_kernel(const float* __restrict__ x, half_t* __restrict__ S) {
  size_t i8 = ((size_t)blockIdx.x * 256 + threadIdx.x) * 8;
  const size_t stride = (size_t)B_ * C_ * HW_;
  float v[8];
#pragma unroll
  for (int j = 0; j < 8; ++j) v[j] = 0.0f;
#pragma unroll
  for (int t = 0; t < T_; ++t) {
    v4f x0 = *(const v4f*)(x + (size_t)t * stride + i8);
    v4f x1 = *(const v4f*)(x + (size_t)t * stride + i8 + 4);
    v8h o;
#pragma unroll
    for (int j = 0; j < 8; ++j) {
      float xv = (j < 4) ? x0[j & 3] : x1[j & 3];
      v[j] += (xv - v[j]) * 0.5f;                  // tau = 2
      float s = (v[j] >= 1.0f) ? 1.0f : 0.0f;
      v[j] -= v[j] * s;                            // hard reset
      o[j] = (half_t)s;
    }
    *(v8h*)(S + (size_t)t * stride + i8) = o;
  }
}

// ---------------- K2/K4: Haar transform with fused NegIF (WMMA) --------------------
// fwd (INV=0):  Z_t = NegIF_t( M @ X_t ) @ M^T      input = spikes (TDM-loaded)
// inv (INV=1):  Z_t = NegIF_t( M^T @ BN2(X_t) ) @ M
// Accumulates per-channel sum/sumsq of Z for training-mode BN.

template <bool INV>
__global__ void __launch_bounds__(128)
haar_kernel(const half_t* __restrict__ Xin, half_t* __restrict__ Zout,
            const half_t* __restrict__ Mg, const half_t* __restrict__ MgT,
            const float* __restrict__ inStats, const float* __restrict__ inG,
            const float* __restrict__ inB, float* __restrict__ statsOut) {
  __shared__ __attribute__((aligned(16))) half_t Msh[64 * 64];
  __shared__ __attribute__((aligned(16))) half_t MshT[64 * 64];
  __shared__ __attribute__((aligned(16))) half_t Ysh[4][64 * 64];
  __shared__ float red[128];
  __shared__ float red2[128];

  const int tid  = threadIdx.x;
  const int lane = tid & 31;
  const int wv   = tid >> 5;
  const int blk  = blockIdx.x;          // B_*C_ = 2048
  const int b    = blk >> 8;
  const int c    = blk & 255;

  float sc = 1.0f, sh = 0.0f;
  if (INV) {
    float mean = inStats[c * 2] * (1.0f / BNCNT);
    float var  = inStats[c * 2 + 1] * (1.0f / BNCNT) - mean * mean;
    sc = inG[c] * rsqrtf(var + EPS_);
    sh = inB[c] - mean * sc;
  }

  // M and M^T: coalesced async b128 copies
  for (int i = tid; i < 512; i += 128) {
    async_ld_b128(&Msh[i * 8], Mg + i * 8);
    async_ld_b128(&MshT[i * 8], MgT + i * 8);
  }
  // X tiles: TDM for raw spike images (fwd), transformed manual loads for inv
  bool tdm_used = false;
  for (int t = 0; t < T_; ++t) {
    size_t off = ((size_t)(t * B_ + b) * C_ + c) * HW_;
    if (!INV) {
#ifdef HAVE_TDM
      if (tid == 0) tdm_copy_row_to_lds(&Ysh[t][0], Xin + off, HW_);
      tdm_used = true;
#else
      for (int i = tid; i < 512; i += 128)
        async_ld_b128(&Ysh[t][i * 8], Xin + off + i * 8);
#endif
    } else {
      for (int i = tid; i < 512; i += 128) {
        v8h xv = *(const v8h*)(Xin + off + i * 8);
        v8h o;
#pragma unroll
        for (int j = 0; j < 8; ++j) o[j] = (half_t)((float)xv[j] * sc + sh);
        *(v8h*)&Ysh[t][i * 8] = o;
      }
    }
  }
#ifdef HAVE_TDM
  if (tdm_used && tid == 0) __builtin_amdgcn_s_wait_tensorcnt(0);
#endif
  (void)tdm_used;
  async_wait();
  __syncthreads();

  // Stage 1: Y_t = Mop1 @ X_t, buffered in regs, written back in-place (f16)
  for (int t = 0; t < T_; ++t) {
    v8f acc[4];
#pragma unroll
    for (int q = 0; q < 4; ++q) {
      int tile = wv * 4 + q;
      int it = tile >> 2, jt = tile & 3;
      v8f d = {};
#pragma unroll
      for (int ks = 0; ks < 2; ++ks) {
        int k0 = ks * 32;
        v16h a = load_a_rm((INV ? MshT : Msh) + (it * 16) * 64 + k0, 64, lane);
        v16h bf = make_b_frag(lane, [&](int k, int n) -> half_t {
          return Ysh[t][(k0 + k) * 64 + jt * 16 + n];
        });
        d = wmma_f16(a, bf, d);
      }
      acc[q] = d;
    }
    __syncthreads();
    {
      int n = lane & 15, mb = (lane >> 4) ? 8 : 0;
#pragma unroll
      for (int q = 0; q < 4; ++q) {
        int tile = wv * 4 + q;
        int it = tile >> 2, jt = tile & 3;
#pragma unroll
        for (int r = 0; r < 8; ++r)
          Ysh[t][(it * 16 + mb + r) * 64 + jt * 16 + n] = (half_t)acc[q][r];
      }
    }
    __syncthreads();
  }

  // NegIF scan across t, per element (spikes in {-1,0,1})
  for (int e = tid; e < HW_; e += 128) {
    float v = 0.0f;
#pragma unroll
    for (int t = 0; t < T_; ++t) {
      v += (float)Ysh[t][e];
      float s = ((v - 1.0f >= 0.0f) ? 1.0f : 0.0f) -
                ((-v - 1.0f >= 0.0f) ? 1.0f : 0.0f);
      v -= s;                                            // vth = 1 subtraction reset
      Ysh[t][e] = (half_t)s;
    }
  }
  __syncthreads();

  // Stage 2: Z_t = V_t @ Mop2 -> staged in LDS -> coalesced b128 store + BN stats
  float psum = 0.0f, psq = 0.0f;
  for (int t = 0; t < T_; ++t) {
    v8f acc[4];
#pragma unroll
    for (int q = 0; q < 4; ++q) {
      int tile = wv * 4 + q;
      int it = tile >> 2, jt = tile & 3;
      v8f d = {};
#pragma unroll
      for (int ks = 0; ks < 2; ++ks) {
        int k0 = ks * 32;
        v16h a = load_a_rm(Ysh[t] + (it * 16) * 64 + k0, 64, lane);
        v16h bf =
            load_b_kcontig((INV ? MshT : Msh) + (jt * 16) * 64 + k0, 64, lane);
        d = wmma_f16(a, bf, d);
      }
      acc[q] = d;
    }
    __syncthreads();
    {
      int n = lane & 15, mb = (lane >> 4) ? 8 : 0;
#pragma unroll
      for (int q = 0; q < 4; ++q) {
        int tile = wv * 4 + q;
        int it = tile >> 2, jt = tile & 3;
#pragma unroll
        for (int r = 0; r < 8; ++r)
          Ysh[t][(it * 16 + mb + r) * 64 + jt * 16 + n] = (half_t)acc[q][r];
      }
    }
    __syncthreads();
    size_t off = ((size_t)(t * B_ + b) * C_ + c) * HW_;
    for (int i = tid; i < 512; i += 128) {
      v8h zv = *(const v8h*)&Ysh[t][i * 8];
      *(v8h*)(Zout + off + i * 8) = zv;
#pragma unroll
      for (int j = 0; j < 8; ++j) {
        float f = (float)zv[j];
        psum += f;
        psq += f * f;
      }
    }
  }

  red[tid] = psum;
  red2[tid] = psq;
  __syncthreads();
  for (int s2 = 64; s2 > 0; s2 >>= 1) {
    if (tid < s2) { red[tid] += red[tid + s2]; red2[tid] += red2[tid + s2]; }
    __syncthreads();
  }
  if (tid == 0) {
    atomicAdd(&statsOut[c * 2 + 0], red[0]);
    atomicAdd(&statsOut[c * 2 + 1], red2[0]);
  }
}

// ---------------- K3: BN1 -> LIF -> block-diagonal 16x16 mix -> BN2 stats ----------

__global__ void __launch_bounds__(64)
mix_kernel(const half_t* __restrict__ H1, half_t* __restrict__ H2,
           const float* __restrict__ hw, const float* __restrict__ g1,
           const float* __restrict__ b1, const float* __restrict__ statsIn,
           float* __restrict__ statsOut) {
  __shared__ float Wm[256];
  __shared__ float st[32];
  const int tid = threadIdx.x;                 // = w
  const int blk = blockIdx.x;                  // B_*H_*NB_ = 8192
  const int nb = blk & 15;
  const int h  = (blk >> 4) & 63;
  const int b  = blk >> 10;

  for (int i = tid; i < 256; i += 64) Wm[i] = hw[nb * 256 + i];
  if (tid < 32) st[tid] = 0.0f;

  float sc[16], sh[16];
#pragma unroll
  for (int d = 0; d < 16; ++d) {
    int c = nb * 16 + d;
    float mean = statsIn[c * 2] * (1.0f / BNCNT);
    float var  = statsIn[c * 2 + 1] * (1.0f / BNCNT) - mean * mean;
    sc[d] = g1[c] * rsqrtf(var + EPS_);
    sh[d] = b1[c] - mean * sc[d];
  }
  __syncthreads();

  float v[16];
#pragma unroll
  for (int d = 0; d < 16; ++d) v[d] = 0.0f;
  float psum[16], psq[16];
#pragma unroll
  for (int k = 0; k < 16; ++k) { psum[k] = 0.0f; psq[k] = 0.0f; }

  const int w = tid;
  for (int t = 0; t < T_; ++t) {
    float s[16];
#pragma unroll
    for (int d = 0; d < 16; ++d) {
      int c = nb * 16 + d;
      size_t idx = (((size_t)(t * B_ + b) * C_ + c) * H_ + h) * W_ + w;
      float xv = (float)H1[idx] * sc[d] + sh[d];
      v[d] += (xv - v[d]) * 0.5f;
      float sp = (v[d] >= 1.0f) ? 1.0f : 0.0f;
      v[d] -= v[d] * sp;
      s[d] = sp;
    }
#pragma unroll
    for (int k = 0; k < 16; ++k) {
      float y = 0.0f;
#pragma unroll
      for (int d = 0; d < 16; ++d) y = fmaf(s[d], Wm[d * 16 + k], y);
      int c = nb * 16 + k;
      size_t idx = (((size_t)(t * B_ + b) * C_ + c) * H_ + h) * W_ + w;
      H2[idx] = (half_t)y;
      psum[k] += y;
      psq[k] += y * y;
    }
  }
#pragma unroll
  for (int k = 0; k < 16; ++k) {
    atomicAdd(&st[k], psum[k]);
    atomicAdd(&st[16 + k], psq[k]);
  }
  __syncthreads();
  if (tid < 16) {
    atomicAdd(&statsOut[(nb * 16 + tid) * 2 + 0], st[tid]);
    atomicAdd(&statsOut[(nb * 16 + tid) * 2 + 1], st[16 + tid]);
  }
}

// ---------------- K5: conv branch as implicit-GEMM WMMA ----------------------------

__global__ void __launch_bounds__(256)
conv_kernel(const half_t* __restrict__ S, half_t* __restrict__ C1o,
            half_t* __restrict__ C2o, const float* __restrict__ w1,
            const float* __restrict__ w2, float* __restrict__ stats4,
            float* __restrict__ stats5) {
  __shared__ __attribute__((aligned(16))) half_t Stile2[18][66][16]; // ch-last, W halo
  __shared__ __attribute__((aligned(16))) half_t Wpair[5][16][32];   // [pair][o][k2]
  __shared__ __attribute__((aligned(16))) half_t W1pad[16][32];      // [o][k2]
  __shared__ float cst[4][16];

  const int tid  = threadIdx.x;
  const int lane = tid & 31;
  const int wv   = tid >> 5;                 // 8 waves
  const int blk  = blockIdx.x;               // T_*B_*NB_ = 512
  const int nb = blk & 15;
  const int b  = (blk >> 4) & 7;
  const int t  = blk >> 7;
  const int cbase = nb * 16;

  for (int i = tid; i < 5 * 16 * 32; i += 256) {
    int p  = i >> 9;
    int n  = (i >> 5) & 15;
    int k2 = i & 31;
    int tap = (k2 < 16) ? 2 * p : 2 * p + 1;
    int ch  = k2 & 15;
    Wpair[p][n][k2] = (half_t)((tap <= 8) ? w2[n * 144 + ch * 9 + tap] : 0.0f);
  }
  for (int i = tid; i < 512; i += 256) {
    int n = i >> 5, k2 = i & 31;
    W1pad[n][k2] = (half_t)((k2 < 16) ? w1[n * 16 + k2] : 0.0f);
  }
  if (tid < 36) {
    int r = tid >> 1, wsel = (tid & 1) ? 65 : 0;
    *(v8h*)&Stile2[r][wsel][0] = zero8();
    *(v8h*)&Stile2[r][wsel][8] = zero8();
  }
  if (tid < 64) (&cst[0][0])[tid] = 0.0f;
  __syncthreads();

  float a1s = 0.0f, a1q = 0.0f, a2s = 0.0f, a2q = 0.0f;
  const int m  = lane & 15;
  const int kh = (lane >> 4) ? 8 : 0;
  const int nn = lane & 15;
  const int kb = (lane >> 4) ? 16 : 0;

  for (int iter = 0; iter < 4; ++iter) {
    const int h0 = iter * 16;
    for (int i = tid; i < 18 * 64; i += 256) {
      int r = i >> 6;
      int w = i & 63;
      int gh = h0 + r - 1;
      v8h lo = zero8(), hi = zero8();
      if ((unsigned)gh < 64u) {
        size_t sb =
            ((size_t)(t * B_ + b) * C_ + cbase) * HW_ + (size_t)gh * W_ + w;
#pragma unroll
        for (int ch = 0; ch < 8; ++ch) lo[ch] = S[sb + (size_t)ch * HW_];
#pragma unroll
        for (int ch = 0; ch < 8; ++ch) hi[ch] = S[sb + (size_t)(ch + 8) * HW_];
      }
      *(v8h*)&Stile2[r][w + 1][0] = lo;
      *(v8h*)&Stile2[r][w + 1][8] = hi;
    }
    __syncthreads();

    for (int gq = 0; gq < 8; ++gq) {
      int g   = wv * 8 + gq;                  // 64 pixel-groups of 16 px
      int row = g >> 2;
      int wb  = (g & 3) * 16;

      v8f d2 = {};
#pragma unroll
      for (int p = 0; p < 5; ++p) {
        const int ta = 2 * p, tb = 2 * p + 1;
        const int dha = ta / 3, dwa = ta % 3;
        v8h alo = *(const v8h*)&Stile2[row + dha][wb + m + dwa][kh];
        v8h ahi;
        if (tb <= 8) {
          const int dhb = tb / 3, dwb = tb % 3;
          ahi = *(const v8h*)&Stile2[row + dhb][wb + m + dwb][kh];
        } else {
          ahi = zero8();
        }
        v8h blo = *(const v8h*)&Wpair[p][nn][kh];
        v8h bhi = *(const v8h*)&Wpair[p][nn][kh + 16];
        d2 = wmma_f16(cat8(alo, ahi), cat8(blo, bhi), d2);
      }

      v8h a1lo = *(const v8h*)&Stile2[row + 1][wb + m + 1][kh];
      v8h b1lo = *(const v8h*)&W1pad[nn][kh];
      v8h b1hi = *(const v8h*)&W1pad[nn][kh + 16];
      v8f d1 = {};
      d1 = wmma_f16(cat8(a1lo, zero8()), cat8(b1lo, b1hi), d1);

      int mb = (lane >> 4) ? 8 : 0;
      size_t obase =
          (((size_t)(t * B_ + b) * C_ + cbase + nn) * H_ + (h0 + row)) * W_ + wb;
      v8h o1, o2;
#pragma unroll
      for (int r = 0; r < 8; ++r) {
        float y1 = d1[r], y2 = d2[r];
        o1[r] = (half_t)y1;
        o2[r] = (half_t)y2;
        a1s += y1; a1q += y1 * y1;
        a2s += y2; a2q += y2 * y2;
      }
      *(v8h*)(C1o + obase + mb) = o1;        // D M-dim is contiguous pixels
      *(v8h*)(C2o + obase + mb) = o2;
    }
    __syncthreads();
  }

  atomicAdd(&cst[0][nn], a1s);
  atomicAdd(&cst[1][nn], a1q);
  atomicAdd(&cst[2][nn], a2s);
  atomicAdd(&cst[3][nn], a2q);
  __syncthreads();
  if (tid < 16) {
    atomicAdd(&stats4[(cbase + tid) * 2 + 0], cst[0][tid]);
    atomicAdd(&stats4[(cbase + tid) * 2 + 1], cst[1][tid]);
  } else if (tid < 32) {
    int k = tid - 16;
    atomicAdd(&stats5[(cbase + k) * 2 + 0], cst[2][k]);
    atomicAdd(&stats5[(cbase + k) * 2 + 1], cst[3][k]);
  }
}

// ---------------- K6: out = BN3(H3) + BN4(C1) + BN5(C2), 8 elems/thread ------------

__global__ void __launch_bounds__(256)
combine_kernel(const half_t* __restrict__ H3, const half_t* __restrict__ C1,
               const half_t* __restrict__ C2, const float* __restrict__ g3,
               const float* __restrict__ b3, const float* __restrict__ g4,
               const float* __restrict__ b4, const float* __restrict__ g5,
               const float* __restrict__ b5, const float* __restrict__ st3,
               const float* __restrict__ st4, const float* __restrict__ st5,
               float* __restrict__ out) {
  size_t i8 = ((size_t)blockIdx.x * 256 + threadIdx.x) * 8;
  int c = (int)((i8 >> 12) & 255);           // 8 contiguous elems share one channel

  float m3 = st3[c * 2] * (1.0f / BNCNT);
  float v3 = st3[c * 2 + 1] * (1.0f / BNCNT) - m3 * m3;
  float s3 = g3[c] * rsqrtf(v3 + EPS_), h3 = b3[c] - m3 * s3;

  float m4 = st4[c * 2] * (1.0f / BNCNT);
  float v4 = st4[c * 2 + 1] * (1.0f / BNCNT) - m4 * m4;
  float s4 = g4[c] * rsqrtf(v4 + EPS_), h4 = b4[c] - m4 * s4;

  float m5 = st5[c * 2] * (1.0f / BNCNT);
  float v5 = st5[c * 2 + 1] * (1.0f / BNCNT) - m5 * m5;
  float s5 = g5[c] * rsqrtf(v5 + EPS_), h5 = b5[c] - m5 * s5;

  v8h a = *(const v8h*)(H3 + i8);
  v8h d1 = *(const v8h*)(C1 + i8);
  v8h d2 = *(const v8h*)(C2 + i8);
  v4f o0, o1;
#pragma unroll
  for (int j = 0; j < 8; ++j) {
    float r = ((float)a[j] * s3 + h3) + ((float)d1[j] * s4 + h4) +
              ((float)d2[j] * s5 + h5);
    if (j < 4) o0[j & 3] = r; else o1[j & 3] = r;
  }
  *(v4f*)(out + i8) = o0;
  *(v4f*)(out + i8 + 4) = o1;
}

// ---------------- host launch ------------------------------------------------------

extern "C" void kernel_launch(void* const* d_in, const int* in_sizes, int n_in,
                              void* d_out, int out_size, void* d_ws, size_t ws_size,
                              hipStream_t stream) {
  const float* x       = (const float*)d_in[0];
  const float* haar_w  = (const float*)d_in[1];
  const float* fb_g    = (const float*)d_in[2];
  const float* fb_b    = (const float*)d_in[3];
  const float* mb_g    = (const float*)d_in[4];
  const float* mb_b    = (const float*)d_in[5];
  const float* ib_g    = (const float*)d_in[6];
  const float* ib_b    = (const float*)d_in[7];
  const float* conv1_w = (const float*)d_in[8];
  // d_in[9]  = conv1_b  (BN-invariant, unused)
  const float* conv2_w = (const float*)d_in[10];
  // d_in[11] = conv2_b  (BN-invariant, unused)
  const float* c1b_g   = (const float*)d_in[12];
  const float* c1b_b   = (const float*)d_in[13];
  const float* c2b_g   = (const float*)d_in[14];
  const float* c2b_b   = (const float*)d_in[15];
  float* out = (float*)d_out;

  char* ws = (char*)d_ws;
  half_t* Mg  = (half_t*)(ws + 0);              // 8 KB
  half_t* MgT = (half_t*)(ws + 8192);           // 8 KB
  float*  ST  = (float*)(ws + 32768);           // 5 * 512 f32 BN accumulators
  half_t* S   = (half_t*)(ws + 65536);          // spikes          (64 MB)
  half_t* A   = S + NEL;                        // H1, later C1    (64 MB)
  half_t* Bb  = A + NEL;                        // H2, later C2    (64 MB)
  half_t* Cc  = Bb + NEL;                       // H3              (64 MB)
  float* ST1 = ST;
  float* ST2 = ST + 512;
  float* ST3 = ST + 1024;
  float* ST4 = ST + 1536;
  float* ST5 = ST + 2048;

  haar_init_kernel<<<1, 256, 0, stream>>>(Mg, MgT, ST);

  lif_kernel<<<(int)((NEL / T_) / (256 * 8)), 256, 0, stream>>>(x, S);

  haar_kernel<false><<<B_ * C_, 128, 0, stream>>>(S, A, Mg, MgT, nullptr, nullptr,
                                                  nullptr, ST1);

  mix_kernel<<<B_ * H_ * NB_, 64, 0, stream>>>(A, Bb, haar_w, fb_g, fb_b, ST1, ST2);

  haar_kernel<true><<<B_ * C_, 128, 0, stream>>>(Bb, Cc, Mg, MgT, ST2, mb_g, mb_b,
                                                 ST3);

  conv_kernel<<<T_ * B_ * NB_, 256, 0, stream>>>(S, A, Bb, conv1_w, conv2_w, ST4,
                                                 ST5);

  combine_kernel<<<(int)(NEL / (256 * 8)), 256, 0, stream>>>(
      Cc, A, Bb, ib_g, ib_b, c1b_g, c1b_b, c2b_g, c2b_b, ST3, ST4, ST5, out);
}
